// DSVTCrossAttention_48722109006385
// MI455X (gfx1250) — matrix operand order repaired and checked by reading, not compiled, not run on hardware
//
#include <hip/hip_runtime.h>

// ---------------------------------------------------------------------------
// DSVT cross-attention block for MI455X (gfx1250), bf16 WMMA everywhere.
// N=92160, D=192, BD=256, H=8, HD=24, FF=768, S=2560, L=36, M=256
// Round 3: fix register spills seen in round 2 asm. 2 row-tiles per wave
// (16 accum VGPRs), interleaved load->WMMA, un-unrolled K loops,
// __launch_bounds__(128). Keeps 2x B-operand reuse over round 1.
// ---------------------------------------------------------------------------

#define kN  92160
#define kD  192
#define kBD 256
#define kH  8
#define kHD 24
#define kFF 768
#define kS  2560
#define kL  36
#define kM  256

typedef __attribute__((ext_vector_type(16))) __bf16 v16bf;
typedef __attribute__((ext_vector_type(8)))  float  v8f;

__device__ __forceinline__ int afrag_k(int lane, int i) {
    // 16-bit A-matrix 16x32 per-lane K mapping (ISA 7.12.2)
    return i + ((i >= 8) ? 8 : 0) + ((lane >= 16) ? 8 : 0);
}
__device__ __forceinline__ int bfrag_k(int lane, int i) {
    // 16-bit B-matrix 32x16 per-lane K mapping (same K order as A fragment)
    return i + ((i >= 8) ? 8 : 0) + ((lane >= 16) ? 8 : 0);
}

__device__ __forceinline__ v8f wmma_bf16(v16bf a, v16bf b, v8f c) {
    return __builtin_amdgcn_wmma_f32_16x16x32_bf16(
        false, a, false, b, (short)0, c, false, false);
}

// A fragment from LDS bf16 row-major (16-row tile at `base`, stride ld)
__device__ __forceinline__ v16bf load_a_lds(const __bf16* base, int ld, int k0, int lane) {
    v16bf a;
    int m = lane & 15;
#pragma unroll
    for (int i = 0; i < 16; ++i)
        a[i] = base[m * ld + k0 + afrag_k(lane, i)];
    return a;
}

// B fragment where memory is W[j][k] (row-major JxK) and out = A * W^T
__device__ __forceinline__ v16bf load_b_wt(const __bf16* W, int Kdim, int j0, int k0, int lane) {
    v16bf b;
    int j = j0 + (lane & 15);
#pragma unroll
    for (int i = 0; i < 16; ++i)
        b[i] = W[j * Kdim + k0 + bfrag_k(lane, i)];
    return b;
}

// B fragment where memory is B[k][n] (row-major KxN, leading dim ldn)
__device__ __forceinline__ v16bf load_b_kn(const __bf16* B, int ldn, int k0, int n0, int lane) {
    v16bf b;
    int n = n0 + (lane & 15);
#pragma unroll
    for (int i = 0; i < 16; ++i)
        b[i] = B[(k0 + bfrag_k(lane, i)) * ldn + n];
    return b;
}

// ---------------------------------------------------------------------------
// prep kernels
// ---------------------------------------------------------------------------
__global__ void f2bf_kernel(const float* __restrict__ in, __bf16* __restrict__ out, int n) {
    int i = blockIdx.x * blockDim.x + threadIdx.x;
    if (i < n) out[i] = (__bf16)in[i];
}

__global__ void addpos_bf_kernel(const float* __restrict__ src, const float* __restrict__ pos,
                                 __bf16* __restrict__ out, int n) {
    int i = blockIdx.x * blockDim.x + threadIdx.x;
    if (i < n) out[i] = (__bf16)(src[i] + pos[i]);
}

// k = (box_feature+box_pos) @ Wk^T + bk ; v = box_feature @ Wv^T + bv
// fp32 scalar (only 25 MFLOP), stored into WMMA-friendly layouts:
//   kT[h][d(32pad)][m]  (KxN for the scores GEMM B operand)
//   vB[h][m][d(32pad)]  (KxN for the ctx GEMM B operand)
__global__ void kv_kernel(const float* __restrict__ boxf, const float* __restrict__ boxp,
                          const float* __restrict__ Wk, const float* __restrict__ bk,
                          const float* __restrict__ Wv, const float* __restrict__ bv,
                          __bf16* __restrict__ kT, __bf16* __restrict__ vB) {
    int idx = blockIdx.x * blockDim.x + threadIdx.x;
    if (idx >= 2 * kM * kD) return;
    int which = idx / (kM * kD);
    int r = idx % (kM * kD);
    int m = r / kD, j = r % kD;
    int h = j / kHD, d = j % kHD;
    if (which == 0) {
        float acc = bk[j];
        for (int b = 0; b < kBD; ++b)
            acc += (boxf[m * kBD + b] + boxp[m * kBD + b]) * Wk[j * kBD + b];
        kT[(h * 32 + d) * kM + m] = (__bf16)acc;
    } else {
        float acc = bv[j];
        for (int b = 0; b < kBD; ++b)
            acc += boxf[m * kBD + b] * Wv[j * kBD + b];
        vB[(h * kM + m) * 32 + d] = (__bf16)acc;
    }
}

// first occurrence of each voxel index (scatter-back rule)
__global__ void fp_min_kernel(const int* __restrict__ inds, int* __restrict__ fp, int n) {
    int i = blockIdx.x * blockDim.x + threadIdx.x;
    if (i < n) atomicMin(&fp[inds[i]], i);
}

// ---------------------------------------------------------------------------
// Q projection: q[i] = qin[voxel_inds[i]] @ Wq^T + bq, bf16 out.
// grid: N/32 blocks, 128 threads. Wave = 3 col-tiles x 2 row-tiles, K=192.
// ---------------------------------------------------------------------------
__global__ void __launch_bounds__(128)
qproj_kernel(const __bf16* __restrict__ qin, const int* __restrict__ inds,
             const __bf16* __restrict__ Wb, const float* __restrict__ bias,
             __bf16* __restrict__ qout) {
    __shared__ __bf16 xa[32 * kD];  // 12 KB
    int rb = blockIdx.x, tid = threadIdx.x;
    for (int e = tid; e < 32 * kD; e += 128) {
        int r = e / kD, c = e % kD;
        xa[e] = qin[inds[rb * 32 + r] * kD + c];
    }
    __syncthreads();
    int lane = tid & 31, wave = tid >> 5;
#pragma unroll 1
    for (int t = 0; t < 3; ++t) {
        int ct = wave * 3 + t;
        v8f c0 = {}, c1 = {};
#pragma unroll 1
        for (int ks = 0; ks < 6; ++ks) {
            v16bf b = load_b_wt(Wb, kD, ct * 16, ks * 32, lane);
            v16bf a = load_a_lds(xa, kD, ks * 32, lane);
            c0 = wmma_bf16(a, b, c0);
            a = load_a_lds(xa + 16 * kD, kD, ks * 32, lane);
            c1 = wmma_bf16(a, b, c1);
        }
        int col = ct * 16 + (lane & 15);
        int row0 = (lane >> 4) * 8;
        float bj = bias[col];
#pragma unroll
        for (int r = 0; r < 8; ++r) {
            qout[(rb * 32 + row0 + r) * kD + col] = (__bf16)(c0[r] + bj);
            qout[(rb * 32 + 16 + row0 + r) * kD + col] = (__bf16)(c1[r] + bj);
        }
    }
}

// ---------------------------------------------------------------------------
// Attention: one block per (2 sets, head). L=36 padded to 48 rows, HD=24
// padded to 32. Each kT/vB B-fragment feeds both sets (2x L2 reuse).
// ---------------------------------------------------------------------------
__global__ void __launch_bounds__(128)
attn_kernel(const __bf16* __restrict__ qg, const __bf16* __restrict__ kT,
            const __bf16* __restrict__ vB, const int* __restrict__ inds,
            const int* __restrict__ vcoords, const int* __restrict__ bcoords,
            __bf16* __restrict__ ctx, unsigned char* __restrict__ rowflag) {
    __shared__ __bf16 qa[2 * 48 * 32];   // 6 KB
    __shared__ float sc[2 * 48 * kM];    // 96 KB
    __shared__ int vb[2 * kL];
    __shared__ int bb[kM];
    int s0 = blockIdx.x * 2, h = blockIdx.y, tid = threadIdx.x;

    if (tid < 2 * kL) vb[tid] = vcoords[inds[s0 * kL + tid] * 4];
    for (int m = tid; m < kM; m += 128) bb[m] = bcoords[m * 4];
    for (int e = tid; e < 2 * 48 * 32; e += 128) {
        int si = e / (48 * 32), r = e % (48 * 32);
        int l = r >> 5, d = r & 31;
        __bf16 val = (__bf16)0.0f;
        if (l < kL && d < kHD) val = qg[((s0 + si) * kL + l) * kD + h * kHD + d];
        qa[e] = val;
    }
    __syncthreads();

    int lane = tid & 31, wave = tid >> 5;
    const float scale = 0.20412414523193154f;  // 1/sqrt(24)
    const __bf16* kTb = kT + h * 32 * kM;

    // scores: 3 row-tiles x 16 col-tiles; one B per tile drives both sets
#pragma unroll 1
    for (int t = wave; t < 48; t += 4) {
        int rt = t >> 4, ctc = t & 15;
        v16bf b = load_b_kn(kTb, kM, 0, ctc * 16, lane);
        v8f c0 = {}, c1 = {};
        {
            v16bf a = load_a_lds(qa + rt * 16 * 32, 32, 0, lane);
            c0 = wmma_bf16(a, b, c0);
            a = load_a_lds(qa + 48 * 32 + rt * 16 * 32, 32, 0, lane);
            c1 = wmma_bf16(a, b, c1);
        }
        int col = ctc * 16 + (lane & 15);
        int row0 = rt * 16 + (lane >> 4) * 8;
        int bbm = bb[col];
#pragma unroll
        for (int r = 0; r < 8; ++r) {
            int row = row0 + r;
            float v0 = 0.0f, v1 = 0.0f;
            if (row < kL) {
                v0 = c0[r] * scale;
                if (vb[row] != bbm) v0 = -1e30f;           // batch mask, set 0
                v1 = c1[r] * scale;
                if (vb[kL + row] != bbm) v1 = -1e30f;      // batch mask, set 1
            }
            sc[row * kM + col] = v0;
            sc[(48 + row) * kM + col] = v1;
        }
    }
    __syncthreads();

    // row-wise softmax over M=256 (thread t owns one (set,row))
    if (tid < 2 * kL) {
        int si = tid / kL, l = tid % kL;
        float* rp = sc + (si * 48 + l) * kM;
        float mx = -1e30f;
        for (int m = 0; m < kM; ++m) mx = fmaxf(mx, rp[m]);
        if (mx <= -5e29f) {  // fully masked -> zero row (matches nan_to_num later)
            for (int m = 0; m < kM; ++m) rp[m] = 0.0f;
            if (h == 0) rowflag[s0 * kL + tid] = 1;
        } else {
            float sum = 0.0f;
            for (int m = 0; m < kM; ++m) { float e = __expf(rp[m] - mx); rp[m] = e; sum += e; }
            float inv = 1.0f / sum;
            for (int m = 0; m < kM; ++m) rp[m] *= inv;
        }
    }
    __syncthreads();

    // ctx = attn @ v : 3 row-tiles x 2 col-tiles, K = 256 (8 steps), 2 sets
    const __bf16* vBb = vB + h * kM * 32;
#pragma unroll 1
    for (int t = wave; t < 6; t += 4) {
        int rt = t >> 1, nt = t & 1;
        v8f c0 = {}, c1 = {};
        int m = lane & 15;
#pragma unroll 1
        for (int ks = 0; ks < 8; ++ks) {
            v16bf b = load_b_kn(vBb, 32, ks * 32, nt * 16, lane);
            v16bf a;
#pragma unroll
            for (int i = 0; i < 16; ++i)
                a[i] = (__bf16)sc[(rt * 16 + m) * kM + ks * 32 + afrag_k(lane, i)];
            c0 = wmma_bf16(a, b, c0);
#pragma unroll
            for (int i = 0; i < 16; ++i)
                a[i] = (__bf16)sc[(48 + rt * 16 + m) * kM + ks * 32 + afrag_k(lane, i)];
            c1 = wmma_bf16(a, b, c1);
        }
        int col = nt * 16 + (lane & 15);
        int row0 = rt * 16 + (lane >> 4) * 8;
        if (col < kHD) {
#pragma unroll
            for (int r = 0; r < 8; ++r) {
                int row = row0 + r;
                if (row < kL) {
                    ctx[((s0 + 0) * kL + row) * kD + h * kHD + col] = (__bf16)c0[r];
                    ctx[((s0 + 1) * kL + row) * kD + h * kHD + col] = (__bf16)c1[r];
                }
            }
        }
    }
}

// ---------------------------------------------------------------------------
// out-proj (with first_pos gather + rowflag zeroing) + residual + LayerNorm1
// 32 rows/block: 2 row-tiles share each Wo fragment.
// ---------------------------------------------------------------------------
__global__ void __launch_bounds__(128)
outproj_ln_kernel(const __bf16* __restrict__ ctx, const __bf16* __restrict__ Wob,
                  const float* __restrict__ bo, const float* __restrict__ src,
                  const int* __restrict__ firstpos,
                  const unsigned char* __restrict__ rowflag,
                  const float* __restrict__ g1, const float* __restrict__ be1,
                  float* __restrict__ xout, __bf16* __restrict__ xbf) {
    __shared__ __bf16 xa[32 * kD];   // 12 KB
    __shared__ float yb[32 * kD];    // 24 KB
    __shared__ int srcrow[32];
    __shared__ unsigned char rz[32];
    int rb = blockIdx.x, tid = threadIdx.x;
    if (tid < 32) {
        int sr = firstpos[rb * 32 + tid];
        if ((unsigned)sr >= (unsigned)kN) sr = 0;
        srcrow[tid] = sr;
        rz[tid] = rowflag[sr];
    }
    __syncthreads();
    for (int e = tid; e < 32 * kD; e += 128)
        xa[e] = ctx[srcrow[e / kD] * kD + (e % kD)];
    __syncthreads();

    int lane = tid & 31, wave = tid >> 5;
#pragma unroll 1
    for (int t = 0; t < 3; ++t) {
        int ct = wave * 3 + t;
        v8f c0 = {}, c1 = {};
#pragma unroll 1
        for (int ks = 0; ks < 6; ++ks) {
            v16bf b = load_b_wt(Wob, kD, ct * 16, ks * 32, lane);
            v16bf a = load_a_lds(xa, kD, ks * 32, lane);
            c0 = wmma_bf16(a, b, c0);
            a = load_a_lds(xa + 16 * kD, kD, ks * 32, lane);
            c1 = wmma_bf16(a, b, c1);
        }
        int col = ct * 16 + (lane & 15);
        int row0 = (lane >> 4) * 8;
        float bj = bo[col];
#pragma unroll
        for (int r = 0; r < 8; ++r) {
            int row = row0 + r;
            float s2 = rz[row] ? 0.0f : (c0[r] + bj);        // nan_to_num -> 0
            yb[row * kD + col] = src[(rb * 32 + row) * kD + col] + s2;
            row = 16 + row0 + r;
            s2 = rz[row] ? 0.0f : (c1[r] + bj);
            yb[row * kD + col] = src[(rb * 32 + row) * kD + col] + s2;
        }
    }
    __syncthreads();

    if (tid < 32) {
        float* rp = yb + tid * kD;
        float mean = 0.0f;
        for (int j = 0; j < kD; ++j) mean += rp[j];
        mean *= (1.0f / kD);
        float var = 0.0f;
        for (int j = 0; j < kD; ++j) { float d = rp[j] - mean; var += d * d; }
        var *= (1.0f / kD);
        float inv = rsqrtf(var + 1e-5f);
        int base = (rb * 32 + tid) * kD;
        for (int j = 0; j < kD; ++j) {
            float xv = (rp[j] - mean) * inv * g1[j] + be1[j];
            xout[base + j] = xv;
            xbf[base + j] = (__bf16)xv;
        }
    }
}

// ---------------------------------------------------------------------------
// fused FFN + residual + LayerNorm2. 32 rows/block, hidden (32x768 bf16)
// entirely in LDS (no 141 MB HBM round-trip). x-stage aliases the LN buffer
// (x dead after phase 1; yb written only after the barrier).
// LDS: 24 KB (union) + 48 KB (hidden) = 72 KB.
// ---------------------------------------------------------------------------
__global__ void __launch_bounds__(128)
ffn_ln_kernel(const __bf16* __restrict__ xbf, const float* __restrict__ xf,
              const __bf16* __restrict__ W1b, const float* __restrict__ b1,
              const __bf16* __restrict__ W2b, const float* __restrict__ b2,
              const float* __restrict__ g2, const float* __restrict__ be2,
              float* __restrict__ out) {
    __shared__ char smem[32 * kD * 4 + 32 * kFF * 2];
    __bf16* xa = (__bf16*)smem;                   // 32x192 bf16 (first 12 KB)
    float*  yb = (float*)smem;                    // 32x192 f32 (24 KB, after phase 1)
    __bf16* hb = (__bf16*)(smem + 32 * kD * 4);   // 32x768 bf16 (48 KB)
    int rb = blockIdx.x, tid = threadIdx.x;
    for (int e = tid; e < 32 * kD; e += 128)
        xa[e] = xbf[(rb * 32 + e / kD) * kD + (e % kD)];
    __syncthreads();

    int lane = tid & 31, wave = tid >> 5;
    // phase 1: hidden = relu(x @ W1^T + b1), 48 col-tiles, K = 192
#pragma unroll 1
    for (int t = 0; t < 12; ++t) {
        int ct = wave * 12 + t;
        v8f c0 = {}, c1 = {};
#pragma unroll 1
        for (int ks = 0; ks < 6; ++ks) {
            v16bf b = load_b_wt(W1b, kD, ct * 16, ks * 32, lane);
            v16bf a = load_a_lds(xa, kD, ks * 32, lane);
            c0 = wmma_bf16(a, b, c0);
            a = load_a_lds(xa + 16 * kD, kD, ks * 32, lane);
            c1 = wmma_bf16(a, b, c1);
        }
        int col = ct * 16 + (lane & 15);
        int row0 = (lane >> 4) * 8;
        float bj = b1[col];
#pragma unroll
        for (int r = 0; r < 8; ++r) {
            hb[(row0 + r) * kFF + col] = (__bf16)fmaxf(c0[r] + bj, 0.0f);
            hb[(16 + row0 + r) * kFF + col] = (__bf16)fmaxf(c1[r] + bj, 0.0f);
        }
    }
    __syncthreads();   // xa fully consumed; yb may now overwrite it

    // phase 2: y = x + hidden @ W2^T + b2, 12 col-tiles, K = 768 (24 steps)
#pragma unroll 1
    for (int t = 0; t < 3; ++t) {
        int ct = wave * 3 + t;
        v8f c0 = {}, c1 = {};
#pragma unroll 1
        for (int ks = 0; ks < 24; ++ks) {
            v16bf b = load_b_wt(W2b, kFF, ct * 16, ks * 32, lane);
            v16bf a = load_a_lds(hb, kFF, ks * 32, lane);
            c0 = wmma_bf16(a, b, c0);
            a = load_a_lds(hb + 16 * kFF, kFF, ks * 32, lane);
            c1 = wmma_bf16(a, b, c1);
        }
        int col = ct * 16 + (lane & 15);
        int row0 = (lane >> 4) * 8;
        float bj = b2[col];
#pragma unroll
        for (int r = 0; r < 8; ++r) {
            int row = row0 + r;
            yb[row * kD + col] = xf[(rb * 32 + row) * kD + col] + c0[r] + bj;
            row = 16 + row0 + r;
            yb[row * kD + col] = xf[(rb * 32 + row) * kD + col] + c1[r] + bj;
        }
    }
    __syncthreads();

    if (tid < 32) {
        float* rp = yb + tid * kD;
        float mean = 0.0f;
        for (int j = 0; j < kD; ++j) mean += rp[j];
        mean *= (1.0f / kD);
        float var = 0.0f;
        for (int j = 0; j < kD; ++j) { float d = rp[j] - mean; var += d * d; }
        var *= (1.0f / kD);
        float inv = rsqrtf(var + 1e-5f);
        int base = (rb * 32 + tid) * kD;
        for (int j = 0; j < kD; ++j)
            out[base + j] = (rp[j] - mean) * inv * g2[j] + be2[j];
    }
}

// ---------------------------------------------------------------------------
extern "C" void kernel_launch(void* const* d_in, const int* in_sizes, int n_in,
                              void* d_out, int out_size, void* d_ws, size_t ws_size,
                              hipStream_t stream) {
    const float* src     = (const float*)d_in[0];
    const float* pos     = (const float*)d_in[1];
    const float* boxf    = (const float*)d_in[2];
    const float* boxp    = (const float*)d_in[3];
    const int*   vcoords = (const int*)d_in[4];
    const int*   bcoords = (const int*)d_in[5];
    const int*   vinds   = (const int*)d_in[6];
    const float* Wq = (const float*)d_in[7];   const float* bq = (const float*)d_in[8];
    const float* Wk = (const float*)d_in[9];   const float* bk = (const float*)d_in[10];
    const float* Wv = (const float*)d_in[11];  const float* bv = (const float*)d_in[12];
    const float* Wo = (const float*)d_in[13];  const float* bo = (const float*)d_in[14];
    const float* W1 = (const float*)d_in[15];  const float* b1 = (const float*)d_in[16];
    const float* W2 = (const float*)d_in[17];  const float* b2 = (const float*)d_in[18];
    const float* g1 = (const float*)d_in[19];  const float* be1 = (const float*)d_in[20];
    const float* g2 = (const float*)d_in[21];  const float* be2 = (const float*)d_in[22];
    float* out = (float*)d_out;

    char* ws = (char*)d_ws;
    size_t off = 0;
    auto take = [&](size_t bytes) {
        char* p = ws + off;
        off = (off + bytes + 255) & ~(size_t)255;
        return p;
    };
    __bf16* qin  = (__bf16*)take((size_t)kN * kD * 2);
    __bf16* qbf  = (__bf16*)take((size_t)kN * kD * 2);
    __bf16* ctxb = (__bf16*)take((size_t)kN * kD * 2);
    __bf16* xbf  = (__bf16*)take((size_t)kN * kD * 2);
    float*  xf   = (float*)take((size_t)kN * kD * 4);
    __bf16* Wqb  = (__bf16*)take((size_t)kD * kD * 2);
    __bf16* Wob  = (__bf16*)take((size_t)kD * kD * 2);
    __bf16* W1b  = (__bf16*)take((size_t)kFF * kD * 2);
    __bf16* W2b  = (__bf16*)take((size_t)kD * kFF * 2);
    __bf16* kT   = (__bf16*)take((size_t)kH * 32 * kM * 2);
    __bf16* vB   = (__bf16*)take((size_t)kH * kM * 32 * 2);
    int*    fp   = (int*)take((size_t)kN * 4);
    unsigned char* rowflag = (unsigned char*)take((size_t)kN);

    // --- prep ---
    f2bf_kernel<<<(kD * kD + 255) / 256, 256, 0, stream>>>(Wq, Wqb, kD * kD);
    f2bf_kernel<<<(kD * kD + 255) / 256, 256, 0, stream>>>(Wo, Wob, kD * kD);
    f2bf_kernel<<<(kFF * kD + 255) / 256, 256, 0, stream>>>(W1, W1b, kFF * kD);
    f2bf_kernel<<<(kD * kFF + 255) / 256, 256, 0, stream>>>(W2, W2b, kD * kFF);
    addpos_bf_kernel<<<(kN * kD + 255) / 256, 256, 0, stream>>>(src, pos, qin, kN * kD);

    hipMemsetAsync(kT, 0, (size_t)kH * 32 * kM * 2, stream);       // zero pads d>=24
    hipMemsetAsync(vB, 0, (size_t)kH * kM * 32 * 2, stream);
    kv_kernel<<<(2 * kM * kD + 127) / 128, 128, 0, stream>>>(boxf, boxp, Wk, bk, Wv, bv, kT, vB);

    hipMemsetAsync(fp, 0x7f, (size_t)kN * 4, stream);              // ~INT_MAX sentinel
    fp_min_kernel<<<(kN + 255) / 256, 256, 0, stream>>>(vinds, fp, kN);
    hipMemsetAsync(rowflag, 0, (size_t)kN, stream);

    // --- main pipeline ---
    qproj_kernel<<<kN / 32, 128, 0, stream>>>(qin, vinds, Wqb, bq, qbf);
    attn_kernel<<<dim3(kS / 2, kH), 128, 0, stream>>>(qbf, kT, vB, vinds, vcoords, bcoords,
                                                      ctxb, rowflag);
    outproj_ln_kernel<<<kN / 32, 128, 0, stream>>>(ctxb, Wob, bo, src, fp, rowflag,
                                                   g1, be1, xf, xbf);
    ffn_ln_kernel<<<kN / 32, 128, 0, stream>>>(xbf, xf, W1b, b1, W2b, b2, g2, be2, out);

    (void)in_sizes; (void)n_in; (void)out_size; (void)ws_size;
}
